// GATLearnableModel_90031104458818
// MI455X (gfx1250) — matrix operand order repaired and checked by reading, not compiled, not run on hardware
//
#include <hip/hip_runtime.h>
#include <hip/hip_bf16.h>
#include <math.h>

#define N_NODES 20000
#define E_EDGES 320000
#define HEADS   4
#define DHEAD   64
#define HID     256
#define EMB     64
#define FIN     128
#define OUTF    8

typedef __attribute__((ext_vector_type(16))) _Float16 v16h;
typedef __attribute__((ext_vector_type(8)))  _Float16 v8h;
typedef __attribute__((ext_vector_type(8)))  float    v8f;

__constant__ int c_dstT[9] = {1, 1, 0, 0, 2, 0, 0, 1, 2};

__device__ __forceinline__ float lrelu(float x) { return x > 0.f ? x : 0.2f * x; }

// float atomic max via monotone int/uint mapping (init must be -inf)
__device__ __forceinline__ void atomicMaxF(float* addr, float v) {
  if (v >= 0.f) atomicMax((int*)addr, __float_as_int(v));
  else          atomicMin((unsigned int*)addr, __float_as_uint(v));
}

// ---------------- conversion / fill kernels ----------------
__global__ void k_f32_to_f16(const float* __restrict__ in, _Float16* __restrict__ out, int n) {
  int i = blockIdx.x * blockDim.x + threadIdx.x;
  if (i < n) out[i] = (_Float16)in[i];
}

// W: nm matrices of [K x Nc] row-major f32 -> Wt: nm matrices of [Nc x K] f16
__global__ void k_convT(const float* __restrict__ W, _Float16* __restrict__ Wt,
                        int nm, int K, int Nc) {
  int i = blockIdx.x * blockDim.x + threadIdx.x;
  int per = K * Nc;
  if (i >= nm * per) return;
  int mat = i / per, rem = i % per;
  int r = rem / Nc, c = rem % Nc;
  Wt[(size_t)mat * per + (size_t)c * K + r] = (_Float16)W[i];
}

__global__ void k_fill(float* __restrict__ p, float v, int n) {
  int i = blockIdx.x * blockDim.x + threadIdx.x;
  if (i < n) p[i] = v;
}

// ---------------- WMMA GEMM: C[M x Nc] = A16[M x K] * Bt16[Nc x K]^T ----------------
// M % 16 == 0, K % 32 == 0, Nc % 64 == 0. One wave -> 16x64 output tile.
__global__ void k_gemm_wmma(const _Float16* __restrict__ A,
                            const _Float16* __restrict__ Bt,
                            float* __restrict__ C,
                            int M, int K, int Nc) {
  const int lane   = threadIdx.x & 31;
  const int wave   = blockIdx.x * (blockDim.x >> 5) + (threadIdx.x >> 5);
  const int tilesN = Nc >> 6;
  const int total  = (M >> 4) * tilesN;
  if (wave >= total) return;                 // wave-uniform: EXEC stays all-ones
  const int tm  = wave / tilesN;
  const int tn  = wave % tilesN;
  const int l15 = lane & 15;
  const int hi  = lane >> 4;

  const _Float16* arow = A + (size_t)(tm * 16 + l15) * K + 16 * hi;
  const _Float16* bcol = Bt + (size_t)(tn * 64 + l15) * K + 16 * hi;

  v8f acc[4] = {};
  for (int k0 = 0; k0 < K; k0 += 32) {
    union { v16h v; v8h h[2]; } a;
    a.h[0] = *(const v8h*)(arow + k0);        // K = k0 + 16*hi     .. +7
    a.h[1] = *(const v8h*)(arow + k0 + 8);    // K = k0 + 16*hi + 8 .. +15
#pragma unroll
    for (int j = 0; j < 4; ++j) {
      v16h b = *(const v16h*)(bcol + (size_t)j * 16 * K + k0);
      acc[j] = __builtin_amdgcn_wmma_f32_16x16x32_f16(
          false, a.v, false, b, (short)0, acc[j], false, false);
    }
  }
#pragma unroll
  for (int j = 0; j < 4; ++j) {
    float* cp = C + (size_t)(tm * 16 + 8 * hi) * Nc + tn * 64 + j * 16 + l15;
#pragma unroll
    for (int r = 0; r < 8; ++r) cp[(size_t)r * Nc] = acc[j][r];
  }
}

// h16 = f16(C + bias[col]) for embedding epilogue
__global__ void k_bias_h16(const float* __restrict__ C, const float* __restrict__ bias,
                           _Float16* __restrict__ out, int Nc, int total) {
  int i = blockIdx.x * blockDim.x + threadIdx.x;
  if (i < total) out[i] = (_Float16)(C[i] + bias[i % Nc]);
}

// el[n,h] = dot(z_src[n,h,:], al[h,:]);  er[n,h] = dot(z_dst[n,h,:], ar[h,:])
__global__ void k_el_er(const float* __restrict__ zs, const float* __restrict__ zd,
                        const float* __restrict__ al, const float* __restrict__ ar,
                        float* __restrict__ el, float* __restrict__ er) {
  int i = blockIdx.x * blockDim.x + threadIdx.x;
  if (i >= N_NODES * HEADS) return;
  int n = i >> 2, h = i & 3;
  const float* z1 = zs + (size_t)n * HID + h * DHEAD;
  const float* z2 = zd + (size_t)n * HID + h * DHEAD;
  const float* a1 = al + h * DHEAD;
  const float* a2 = ar + h * DHEAD;
  float s1 = 0.f, s2 = 0.f;
#pragma unroll 8
  for (int d = 0; d < DHEAD; ++d) { s1 += z1[d] * a1[d]; s2 += z2[d] * a2[d]; }
  el[i] = s1; er[i] = s2;
}

__global__ void k_edge_max(const int* __restrict__ src, const int* __restrict__ dst,
                           const float* __restrict__ el, const float* __restrict__ er,
                           float* __restrict__ m) {
  int i = blockIdx.x * blockDim.x + threadIdx.x;
  if (i >= E_EDGES) return;
  int s = src[i], d = dst[i];
  float4 ev = ((const float4*)el)[s];
  float4 rv = ((const float4*)er)[d];
  float* mp = m + (size_t)d * 4;
  atomicMaxF(mp + 0, lrelu(ev.x + rv.x));
  atomicMaxF(mp + 1, lrelu(ev.y + rv.y));
  atomicMaxF(mp + 2, lrelu(ev.z + rv.z));
  atomicMaxF(mp + 3, lrelu(ev.w + rv.w));
}

__global__ void k_edge_den(const int* __restrict__ src, const int* __restrict__ dst,
                           const float* __restrict__ el, const float* __restrict__ er,
                           const float* __restrict__ m, float* __restrict__ den) {
  int i = blockIdx.x * blockDim.x + threadIdx.x;
  if (i >= E_EDGES) return;
  int s = src[i], d = dst[i];
  float4 ev = ((const float4*)el)[s];
  float4 rv = ((const float4*)er)[d];
  float4 mv = ((const float4*)m)[d];
  float* dp = den + (size_t)d * 4;
  atomicAdd(dp + 0, __expf(lrelu(ev.x + rv.x) - mv.x));
  atomicAdd(dp + 1, __expf(lrelu(ev.y + rv.y) - mv.y));
  atomicAdd(dp + 2, __expf(lrelu(ev.z + rv.z) - mv.z));
  atomicAdd(dp + 3, __expf(lrelu(ev.w + rv.w) - mv.w));
}

// 64 threads per edge: out[dst] += alpha[h] * z_src[src]  (256 floats)
__global__ void k_edge_agg(const int* __restrict__ src, const int* __restrict__ dst,
                           const float* __restrict__ el, const float* __restrict__ er,
                           const float* __restrict__ m, const float* __restrict__ den,
                           const float* __restrict__ zsrc, float* __restrict__ outdt) {
  int t    = threadIdx.x & 63;
  int eidx = blockIdx.x * (blockDim.x >> 6) + (threadIdx.x >> 6);
  if (eidx >= E_EDGES) return;
  int s = src[eidx], d = dst[eidx];
  float4 ev = ((const float4*)el)[s];
  float4 rv = ((const float4*)er)[d];
  float4 mv = ((const float4*)m)[d];
  float4 dv = ((const float4*)den)[d];
  float a0 = __expf(lrelu(ev.x + rv.x) - mv.x) / dv.x;
  float a1 = __expf(lrelu(ev.y + rv.y) - mv.y) / dv.y;
  float a2 = __expf(lrelu(ev.z + rv.z) - mv.z) / dv.z;
  float a3 = __expf(lrelu(ev.w + rv.w) - mv.w) / dv.w;
  const float* zp = zsrc  + (size_t)s * HID;
  float*       op = outdt + (size_t)d * HID;
  atomicAdd(op + 0 * 64 + t, a0 * zp[0 * 64 + t]);
  atomicAdd(op + 1 * 64 + t, a1 * zp[1 * 64 + t]);
  atomicAdd(op + 2 * 64 + t, a2 * zp[2 * 64 + t]);
  atomicAdd(op + 3 * 64 + t, a3 * zp[3 * 64 + t]);
}

// per-node LayerNorm + ReLU; adds per-edge-type bias (reference adds b[e] to all dst-type nodes)
__global__ void k_ln_relu(const float* __restrict__ acc, const float* __restrict__ b9,
                          const float* __restrict__ gamma, const float* __restrict__ beta,
                          _Float16* __restrict__ hout) {
  int node = blockIdx.x;                // 0 .. 3N-1
  int t = node / N_NODES;
  int c = threadIdx.x;                  // 0 .. 255
  float v = acc[(size_t)node * HID + c];
#pragma unroll
  for (int e = 0; e < 9; ++e)
    if (c_dstT[e] == t) v += b9[e * HID + c];
  __shared__ float ssum[256], ssq[256];
  ssum[c] = v; ssq[c] = v * v;
  __syncthreads();
  for (int st = 128; st > 0; st >>= 1) {
    if (c < st) { ssum[c] += ssum[c + st]; ssq[c] += ssq[c + st]; }
    __syncthreads();
  }
  float mu  = ssum[0] * (1.f / HID);
  float var = ssq[0] * (1.f / HID) - mu * mu;
  float y = (v - mu) * rsqrtf(var + 1e-5f) * gamma[t * HID + c] + beta[t * HID + c];
  hout[(size_t)node * HID + c] = (_Float16)fmaxf(y, 0.f);
}

__global__ void k_classify(const _Float16* __restrict__ h16, const float* __restrict__ Wc,
                           const float* __restrict__ bc, float* __restrict__ out) {
  int i = blockIdx.x * blockDim.x + threadIdx.x;
  if (i >= 3 * N_NODES) return;
  int t = i / N_NODES;
  const _Float16* hp = h16 + (size_t)i * HID;
  const float* wc = Wc + (size_t)t * HID * OUTF;
  float acc[OUTF];
#pragma unroll
  for (int o = 0; o < OUTF; ++o) acc[o] = bc[t * OUTF + o];
  for (int c = 0; c < HID; ++c) {
    float hv = (float)hp[c];
    const float* w = wc + c * OUTF;
#pragma unroll
    for (int o = 0; o < OUTF; ++o) acc[o] += hv * w[o];
  }
#pragma unroll
  for (int o = 0; o < OUTF; ++o) out[(size_t)i * OUTF + o] = acc[o];
}

// ---------------- host orchestration ----------------
static inline int cdiv(long a, long b) { return (int)((a + b - 1) / b); }

extern "C" void kernel_launch(void* const* d_in, const int* in_sizes, int n_in,
                              void* d_out, int out_size, void* d_ws, size_t ws_size,
                              hipStream_t stream) {
  const float* x     = (const float*)d_in[0];
  const int*   edges = (const int*)  d_in[1];
  const float* Wemb  = (const float*)d_in[2];
  const float* bemb  = (const float*)d_in[3];
  const float* W0    = (const float*)d_in[4];
  const float* al0   = (const float*)d_in[5];
  const float* ar0   = (const float*)d_in[6];
  const float* b0    = (const float*)d_in[7];
  const float* W1    = (const float*)d_in[8];
  const float* al1   = (const float*)d_in[9];
  const float* ar1   = (const float*)d_in[10];
  const float* b1    = (const float*)d_in[11];
  const float* gamma = (const float*)d_in[12];
  const float* beta  = (const float*)d_in[13];
  const float* Wc    = (const float*)d_in[14];
  const float* bc    = (const float*)d_in[15];
  float* out = (float*)d_out;

  static const int SRC_T[9] = {0, 2, 2, 0, 1, 1, 0, 1, 2};
  static const int DST_T[9] = {1, 1, 0, 0, 2, 0, 0, 1, 2};

  // workspace carve (256B aligned)
  size_t off = 0;
  char* base = (char*)d_ws;
  auto take = [&](size_t bytes) -> char* {
    char* p = base + off;
    off += (bytes + 255) & ~(size_t)255;
    return p;
  };
  _Float16* x16    = (_Float16*)take((size_t)3 * N_NODES * FIN * 2);
  _Float16* h16    = (_Float16*)take((size_t)3 * N_NODES * HID * 2);
  _Float16* wembT  = (_Float16*)take((size_t)3 * EMB * FIN * 2);
  _Float16* w0T    = (_Float16*)take((size_t)9 * HID * EMB * 2);
  _Float16* w1T    = (_Float16*)take((size_t)18 * HID * HID * 2);
  float*    zsrc   = (float*)take((size_t)N_NODES * HID * 4);
  float*    zdst   = (float*)take((size_t)N_NODES * HID * 4);
  float*    elb    = (float*)take((size_t)N_NODES * HEADS * 4);
  float*    erb    = (float*)take((size_t)N_NODES * HEADS * 4);
  float*    mb     = (float*)take((size_t)N_NODES * HEADS * 4);
  float*    denb   = (float*)take((size_t)N_NODES * HEADS * 4);
  float*    outacc = (float*)take((size_t)3 * N_NODES * HID * 4);
  (void)ws_size; (void)in_sizes; (void)n_in; (void)out_size;

  const int TB = 256;
  // precision conversions / weight transposes
  k_f32_to_f16<<<cdiv((long)3 * N_NODES * FIN, TB), TB, 0, stream>>>(x, x16, 3 * N_NODES * FIN);
  k_convT<<<cdiv((long)3 * FIN * EMB, TB), TB, 0, stream>>>(Wemb, wembT, 3, FIN, EMB);
  k_convT<<<cdiv((long)9 * EMB * HID, TB), TB, 0, stream>>>(W0, w0T, 9, EMB, HID);
  k_convT<<<cdiv((long)18 * HID * HID, TB), TB, 0, stream>>>(W1, w1T, 18, HID, HID);

  // embedding: h[t] = x[t] @ Wemb[t] + bemb[t]   (stored f16, stride EMB)
  for (int t = 0; t < 3; ++t) {
    int waves = (N_NODES / 16) * (EMB / 64);
    k_gemm_wmma<<<cdiv(waves, 8), TB, 0, stream>>>(
        x16 + (size_t)t * N_NODES * FIN, wembT + (size_t)t * EMB * FIN, zsrc,
        N_NODES, FIN, EMB);
    k_bias_h16<<<cdiv((long)N_NODES * EMB, TB), TB, 0, stream>>>(
        zsrc, bemb + t * EMB, h16 + (size_t)t * N_NODES * EMB, EMB, N_NODES * EMB);
  }

  int Din = EMB;
  for (int l = 0; l < 3; ++l) {
    const _Float16* WtL;
    const float *alL, *arL, *bL;
    if (l == 0) { WtL = w0T; alL = al0; arL = ar0; bL = b0; }
    else {
      WtL = w1T + (size_t)(l - 1) * 9 * HID * HID;
      alL = al1 + (size_t)(l - 1) * 9 * HEADS * DHEAD;
      arL = ar1 + (size_t)(l - 1) * 9 * HEADS * DHEAD;
      bL  = b1  + (size_t)(l - 1) * 9 * HID;
    }
    k_fill<<<cdiv((long)3 * N_NODES * HID, TB), TB, 0, stream>>>(outacc, 0.f, 3 * N_NODES * HID);

    int waves = (N_NODES / 16) * (HID / 64);
    for (int e = 0; e < 9; ++e) {
      int st = SRC_T[e], dt = DST_T[e];
      const _Float16* We = WtL + (size_t)e * HID * Din;
      k_gemm_wmma<<<cdiv(waves, 8), TB, 0, stream>>>(
          h16 + (size_t)st * N_NODES * Din, We, zsrc, N_NODES, Din, HID);
      const float* zd = zsrc;
      if (st != dt) {
        k_gemm_wmma<<<cdiv(waves, 8), TB, 0, stream>>>(
            h16 + (size_t)dt * N_NODES * Din, We, zdst, N_NODES, Din, HID);
        zd = zdst;
      }
      k_el_er<<<cdiv((long)N_NODES * HEADS, TB), TB, 0, stream>>>(
          zsrc, zd, alL + e * HEADS * DHEAD, arL + e * HEADS * DHEAD, elb, erb);
      k_fill<<<cdiv((long)N_NODES * HEADS, TB), TB, 0, stream>>>(mb, -INFINITY, N_NODES * HEADS);
      k_fill<<<cdiv((long)N_NODES * HEADS, TB), TB, 0, stream>>>(denb, 0.f, N_NODES * HEADS);
      const int* srcI = edges + (size_t)e * 2 * E_EDGES;
      const int* dstI = srcI + E_EDGES;
      k_edge_max<<<cdiv(E_EDGES, TB), TB, 0, stream>>>(srcI, dstI, elb, erb, mb);
      k_edge_den<<<cdiv(E_EDGES, TB), TB, 0, stream>>>(srcI, dstI, elb, erb, mb, denb);
      k_edge_agg<<<cdiv(E_EDGES, 4), TB, 0, stream>>>(
          srcI, dstI, elb, erb, mb, denb, zsrc, outacc + (size_t)dt * N_NODES * HID);
    }
    k_ln_relu<<<3 * N_NODES, HID, 0, stream>>>(outacc, bL, gamma, beta, h16);
    Din = HID;
  }

  k_classify<<<cdiv((long)3 * N_NODES, TB), TB, 0, stream>>>(h16, Wc, bc, out);
}
